// BitLinear_13503377178821
// MI455X (gfx1250) — 8-bit matrix operand order repaired and checked by reading, compiled, not run on hardware
//
#include <hip/hip_runtime.h>
#include <hip/hip_bf16.h>

#define N_ROWS 8192
#define IN_F   2048
#define OUT_F  2048

typedef __attribute__((ext_vector_type(16))) int   v16i;
typedef __attribute__((ext_vector_type(4)))  int   v4i;
typedef __attribute__((ext_vector_type(8)))  float v8f;

#if __has_builtin(__builtin_amdgcn_global_load_async_to_lds_b128)
#define HAVE_ASYNC_LDS 1
typedef __attribute__((address_space(1))) v4i g_v4i;
typedef __attribute__((address_space(3))) v4i l_v4i;
#define ASYNC_CP16(dst, src) \
    __builtin_amdgcn_global_load_async_to_lds_b128((g_v4i*)(src), (l_v4i*)(dst), 0, 0)
#else
#define HAVE_ASYNC_LDS 0
#define ASYNC_CP16(dst, src) (*(float4*)(dst) = *(const float4*)(src))
#endif

#if __has_builtin(__builtin_amdgcn_s_wait_asynccnt)
#define WAIT_ASYNC(n) __builtin_amdgcn_s_wait_asynccnt(n)
#else
#define WAIT_ASYNC(n) asm volatile("s_wait_asynccnt %0" :: "n"(n) : "memory")
#endif


__device__ __forceinline__ unsigned char f32_to_e4m3(float f) {
    unsigned int u = __float_as_uint(f);
    unsigned char sign = (unsigned char)((u >> 24) & 0x80u);
    float a = fabsf(f);
    if (!(a > 0.f)) return sign;
    if (a >= 448.f) return (unsigned char)(sign | 0x7E);
    int e = (int)((u >> 23) & 0xFF) - 127;
    unsigned char out;
    if (e < -6) {
        int m = (int)rintf(a * 512.f);
        if (m > 8) m = 8;
        if (m == 8) out = 0x08;
        else out = (unsigned char)m;
    } else {
        int q = (int)rintf(a * exp2f((float)(3 - e)));
        if (q == 16) { q = 8; e += 1; }
        if (e > 8) return (unsigned char)(sign | 0x7E);
        out = (unsigned char)(((e + 7) << 3) | (q - 8));
    }
    return (unsigned char)(sign | out);
}


__global__ void init_qs_kernel(float* qs) {
    int i = blockIdx.x * blockDim.x + threadIdx.x;
    if (i < N_ROWS) qs[i] = 0.f;
}

__global__ void gamma_fp8_kernel(const float* __restrict__ gamma,
                                 unsigned char* __restrict__ g8) {
    int k = blockIdx.x * blockDim.x + threadIdx.x;
    if (k < IN_F) g8[k] = f32_to_e4m3(1.f / gamma[k]);
}

__global__ void quantA_kernel(const float* __restrict__ input,
                              const unsigned char* __restrict__ g8,
                              unsigned char* __restrict__ A8,
                              float* __restrict__ qs) {
    const int mt = blockIdx.x;
    const int kt = blockIdx.y;
    const int tid = threadIdx.x;
    const int lane = tid >> 4;
    const int v    = tid & 15;
    const int m    = lane & 15;
    const int h   = v >> 3;
    const int vp  = v & 7;
    const int g   = vp >> 1;
    const int sub = vp & 1;
    const int kb  = (h << 6) + (g << 4) + ((lane >> 4) << 3) + (sub << 2);

    const int n = mt * 16 + m;
    const int kglob = kt * 128 + kb;

    const float4 x = *(const float4*)(input + (size_t)n * IN_F + kglob);
    const unsigned int gw = *(const unsigned int*)(g8 + kglob);

    unsigned int word = 0;
    float amax;
    {
        const float xs[4] = {x.x, x.y, x.z, x.w};
        amax = fabsf(xs[0]);
        #pragma unroll
        for (int b = 0; b < 4; ++b) {
            float xv = xs[b];
            unsigned int gb = (gw >> (8 * b)) & 0xFFu;
            unsigned int byte = (xv > 0.f) ? gb : ((xv < 0.f) ? (gb | 0x80u) : 0u);
            word |= byte << (8 * b);
            amax = fmaxf(amax, fabsf(xv));
        }
    }

    unsigned int* dst = (unsigned int*)(A8 + ((size_t)(mt * 16 + kt)) * 2048);
    dst[tid] = word;

    __shared__ int smax[16];
    if (tid < 16) smax[tid] = 0;
    __syncthreads();
    atomicMax(&smax[m], __float_as_int(amax));
    __syncthreads();
    if (tid < 16) atomicMax((int*)&qs[mt * 16 + tid], smax[tid]);
}

__global__ void quantW_kernel(const float* __restrict__ weight,
                              unsigned char* __restrict__ W8) {
    const int nt = blockIdx.x;
    const int kt = blockIdx.y;
    const int tid = threadIdx.x;
    const int lane = tid >> 4;
    const int v    = tid & 15;
    const int col  = lane & 15;
    const int kb = ((v >> 3) << 6) + (((v >> 1) & 3) << 4) + ((lane >> 4) << 3) + ((v & 1) << 2);

    const int o = nt * 16 + col;
    const int kglob = kt * 128 + kb;
    const float4 w = *(const float4*)(weight + (size_t)o * IN_F + kglob);

    const float ws[4] = {w.x, w.y, w.z, w.w};
    unsigned int word = 0;
    #pragma unroll
    for (int b = 0; b < 4; ++b) {
        unsigned int byte = (ws[b] > 0.f) ? 0x38u : ((ws[b] < 0.f) ? 0xB8u : 0u);
        word |= byte << (8 * b);
    }
    unsigned int* dst = (unsigned int*)(W8 + ((size_t)(nt * 16 + kt)) * 2048);
    dst[tid] = word;
}

__global__ void __launch_bounds__(512)
gemm_fp8_kernel(const unsigned char* __restrict__ A8,
                const unsigned char* __restrict__ W8,
                const float* __restrict__ qs,
                const float* __restrict__ bias,
                const float* __restrict__ beta,
                float* __restrict__ out) {
    __shared__ __align__(16) unsigned char sbuf[2][32768];

    const int tid  = threadIdx.x;
    const int wv   = tid >> 5;
    const int lane = tid & 31;
    const int mi   = wv >> 1;
    const int nj   = wv & 1;
    const int bm   = blockIdx.x;
    const int bn   = blockIdx.y;

    const int t0  = tid >> 7;
    const int off = (tid & 127) * 16;
    const unsigned char* srcA = A8 + ((size_t)(bm * 8 + t0) * 16) * 2048 + off;
    const unsigned char* srcB = W8 + ((size_t)(bn * 8 + t0) * 16) * 2048 + off;
    unsigned char* dstA = &sbuf[0][t0 * 2048 + off];
    unsigned char* dstB = &sbuf[0][16384 + t0 * 2048 + off];

    #define FILL(buf, kt)                                                      \
        do {                                                                   \
            ASYNC_CP16(dstA + (buf) * 32768,        srcA + (kt) * 2048);       \
            ASYNC_CP16(dstA + (buf) * 32768 + 8192, srcA + (kt) * 2048 + 131072); \
            ASYNC_CP16(dstB + (buf) * 32768,        srcB + (kt) * 2048);       \
            ASYNC_CP16(dstB + (buf) * 32768 + 8192, srcB + (kt) * 2048 + 131072); \
        } while (0)

    v8f acc0 = {}, acc1 = {}, acc2 = {}, acc3 = {};

    const unsigned char* As0 = &sbuf[0][0]     + mi * 2048 + lane * 64;
    const unsigned char* Bs0 = &sbuf[0][16384] + nj * 8192 + lane * 64;

    FILL(0, 0);
    #pragma unroll
    for (int kt = 0; kt < 16; ++kt) {
        const int cur = kt & 1;
#if HAVE_ASYNC_LDS
        WAIT_ASYNC(0);
#endif
        __syncthreads();

        if (kt < 15) FILL(cur ^ 1, kt + 1);

        const unsigned char* As = As0 + cur * 32768;
        const unsigned char* Bs = Bs0 + cur * 32768;
        const v16i a  = *(const v16i*)(As);
        const v16i b0 = *(const v16i*)(Bs);
        const v16i b1 = *(const v16i*)(Bs + 2048);
        const v16i b2 = *(const v16i*)(Bs + 4096);
        const v16i b3 = *(const v16i*)(Bs + 6144);
        acc0 = __builtin_amdgcn_wmma_f32_16x16x128_fp8_fp8(a, b0, (short)0, acc0, false, false);
        acc1 = __builtin_amdgcn_wmma_f32_16x16x128_fp8_fp8(a, b1, (short)0, acc1, false, false);
        acc2 = __builtin_amdgcn_wmma_f32_16x16x128_fp8_fp8(a, b2, (short)0, acc2, false, false);
        acc3 = __builtin_amdgcn_wmma_f32_16x16x128_fp8_fp8(a, b3, (short)0, acc3, false, false);
    }
    #undef FILL

    const int half = lane >> 4;
    const int colL = lane & 15;
    const int nbase = (bm * 8 + mi) * 16 + half * 8;
    float qsv[8];
    #pragma unroll
    for (int r = 0; r < 8; ++r) qsv[r] = qs[nbase + r];

    v8f accs[4] = {acc0, acc1, acc2, acc3};
    #pragma unroll
    for (int s = 0; s < 4; ++s) {
        const int o = bn * 128 + (nj * 4 + s) * 16 + colL;
        const float bi = bias[o];
        const float be = beta[o];
        #pragma unroll
        for (int r = 0; r < 8; ++r) {
            out[(size_t)(nbase + r) * OUT_F + o] = (qsv[r] * accs[s][r] + bi) * be;
        }
    }
}


extern "C" void kernel_launch(void* const* d_in, const int* in_sizes, int n_in,
                              void* d_out, int out_size, void* d_ws, size_t ws_size,
                              hipStream_t stream) {
    const float* input  = (const float*)d_in[0];
    const float* weight = (const float*)d_in[1];
    const float* bias   = (const float*)d_in[2];
    const float* gamma  = (const float*)d_in[3];
    const float* beta   = (const float*)d_in[4];
    float* out = (float*)d_out;

    unsigned char* ws = (unsigned char*)d_ws;
    unsigned char* A8 = ws;
    unsigned char* W8 = ws + (size_t)N_ROWS * IN_F;
    float* qs         = (float*)(W8 + (size_t)OUT_F * IN_F);
    unsigned char* g8 = (unsigned char*)(qs + N_ROWS);
    (void)in_sizes; (void)n_in; (void)out_size; (void)ws_size;

    init_qs_kernel<<<N_ROWS / 256, 256, 0, stream>>>(qs);
    gamma_fp8_kernel<<<IN_F / 256, 256, 0, stream>>>(gamma, g8);
    quantA_kernel<<<dim3(N_ROWS / 16, IN_F / 128), 512, 0, stream>>>(input, g8, A8, qs);
    quantW_kernel<<<dim3(OUT_F / 16, IN_F / 128), 512, 0, stream>>>(weight, W8);
    gemm_fp8_kernel<<<dim3(N_ROWS / 128, OUT_F / 128), 512, 0, stream>>>(A8, W8, qs, bias, beta, out);
}
